// stae_predict_3298534883357
// MI455X (gfx1250) — matrix operand order repaired and checked by reading, not compiled, not run on hardware
//
#include <hip/hip_runtime.h>

// MI455X (gfx1250, wave32) implementation.
// Batched LSTM gates via V_WMMA_F32_16X16X32_F16: M=gate-unit(16/gate-type),
// N=batch(16 per wave), K=[input(4) | h_lo(8) | bias(1) | pad | h_hi(8) | pad].
// K layout chosen so the WMMA D-layout of h feeds the next step's B-layout
// without any cross-lane movement; B construction is branchless (no EXEC
// manipulation inside the serial recurrence chain).

typedef __attribute__((ext_vector_type(16))) _Float16 v16h;
typedef __attribute__((ext_vector_type(8)))  float    v8f;

#define NWAVE 8    // waves per block
#define BPW   16   // batch elements per wave

__device__ __forceinline__ v8f wmma_f16(v16h a, v16h b, v8f c) {
  // D = A(16x32 f16) x B(32x16 f16) + C(16x16 f32)
  return __builtin_amdgcn_wmma_f32_16x16x32_f16(false, a, false, b, (short)0, c,
                                                false, false);
}

__device__ __forceinline__ float tanh_f(float x) {
#if __has_builtin(__builtin_amdgcn_tanhf)
  return __builtin_amdgcn_tanhf(x);      // native v_tanh_f32 on gfx1250
#else
  float e = __expf(2.0f * x);
  return 1.0f - 2.0f * __builtin_amdgcn_rcpf(e + 1.0f);
#endif
}
__device__ __forceinline__ float sigm(float x) {
  return __builtin_fmaf(0.5f, tanh_f(0.5f * x), 0.5f);
}

// Combined-weight element at (gate g, K): K0..3 = W_ih, K4..11 = W_hh[:,0..7],
// K12 = b_ih+b_hh (B supplies 1.0 there), K16..23 = W_hh[:,8..15], else 0.
__device__ __forceinline__ float wcat(const float* __restrict__ Wih,
                                      const float* __restrict__ Whh,
                                      const float* __restrict__ bih,
                                      const float* __restrict__ bhh,
                                      int g, int K) {
  if (K < 4)                 return Wih[g * 4 + K];
  if (K < 12)                return Whh[g * 16 + (K - 4)];
  if (K == 12)               return bih[g] + bhh[g];
  if (K >= 16 && K < 24)     return Whh[g * 16 + (K - 8)];
  return 0.0f;
}

// Build the four 16x32 f16 A tiles (gate blocks i,f,g,o) in contiguous k order (K = 16*hi + e):
// lanes 0-15: M=lane, K in {0..15}; lanes 16-31: M=lane-16, K in {16..31}.
__device__ __forceinline__ void build_A(const float* __restrict__ Wih,
                                        const float* __restrict__ Whh,
                                        const float* __restrict__ bih,
                                        const float* __restrict__ bhh,
                                        int lane, v16h A[4]) {
  const int  m  = lane & 15;
  const bool hi = lane >= 16;
  #pragma unroll
  for (int q = 0; q < 4; ++q) {
    const int g = q * 16 + m;
    v16h a;
    #pragma unroll
    for (int e = 0; e < 16; ++e) {
      const int K = hi ? e + 16
                       : e;
      a[e] = (_Float16)wcat(Wih, Whh, bih, bhh, g, K);
    }
    A[q] = a;
  }
}

// Build B (32x16 f16, contiguous k order), branchless. Lanes 0-15 hold K=0..15
// (inputs f0..f3 at K0..3, h0..7 at K4..11, 1.0 at K12); lanes 16-31 hold
// K=16..31 (h8..15 at K16..23). hr[] is this lane's slice of the previous D
// (hidden units u = 8*hi + r), so no cross-lane movement is needed.
__device__ __forceinline__ v16h build_B(bool hi, float f0, float f1, float f2,
                                        float f3, const float hr[8]) {
  _Float16 h16[8];
  #pragma unroll
  for (int r = 0; r < 8; ++r) h16[r] = (_Float16)hr[r];
  const _Float16 zz = (_Float16)0.0f;
  v16h b;
  b[0]  = hi ? h16[0] : (_Float16)f0;
  b[1]  = hi ? h16[1] : (_Float16)f1;
  b[2]  = hi ? h16[2] : (_Float16)f2;
  b[3]  = hi ? h16[3] : (_Float16)f3;
  b[4]  = hi ? h16[4] : h16[0];
  b[5]  = hi ? h16[5] : h16[1];
  b[6]  = hi ? h16[6] : h16[2];
  b[7]  = hi ? h16[7] : h16[3];
  b[8]  = hi ? zz : h16[4];
  b[9]  = hi ? zz : h16[5];
  b[10] = hi ? zz : h16[6];
  b[11] = hi ? zz : h16[7];
  b[12] = hi ? zz : (_Float16)1.0f;     // bias column
  b[13] = zz;
  b[14] = zz;
  b[15] = zz;
  return b;
}

__device__ __forceinline__ void lstm_step(const v16h A[4], v16h bv,
                                          float cc[8], float hh[8]) {
  const v8f z0 = {0.f, 0.f, 0.f, 0.f, 0.f, 0.f, 0.f, 0.f};
  v8f gi = wmma_f16(A[0], bv, z0);
  v8f gf = wmma_f16(A[1], bv, z0);
  v8f gg = wmma_f16(A[2], bv, z0);
  v8f go = wmma_f16(A[3], bv, z0);
  #pragma unroll
  for (int r = 0; r < 8; ++r) {
    float c = sigm(gf[r]) * cc[r] + sigm(gi[r]) * tanh_f(gg[r]);
    cc[r] = c;
    hh[r] = sigm(go[r]) * tanh_f(c);
  }
}

// order-2 diffusion GCN on 4-node path graph (A symmetric)
__device__ __forceinline__ void gcn4_dev(const float v[4], float w0, float w1,
                                         float w2, float bias, float out[4]) {
  const float x10 = v[1], x11 = v[0] + v[2], x12 = v[1] + v[3], x13 = v[2];
  const float x20 = x11,  x21 = x10 + x12,   x22 = x11 + x13,   x23 = x12;
  out[0] = w0 * v[0] + w1 * x10 + w2 * x20 + bias;
  out[1] = w0 * v[1] + w1 * x11 + w2 * x21 + bias;
  out[2] = w0 * v[2] + w1 * x12 + w2 * x22 + bias;
  out[3] = w0 * v[3] + w1 * x13 + w2 * x23 + bias;
}

__global__ __launch_bounds__(256) void stae_predict_kernel(
    const float* __restrict__ x,    const float* __restrict__ z,
    const float* __restrict__ Wih,  const float* __restrict__ Whh,
    const float* __restrict__ bih,  const float* __restrict__ bhh,
    const float* __restrict__ mW1,  const float* __restrict__ mb1,
    const float* __restrict__ mW2,  const float* __restrict__ mb2,
    const float* __restrict__ Wg,   const float* __restrict__ bg,
    const float* __restrict__ Wg2,  const float* __restrict__ bg2,
    const float* __restrict__ Wl,   const float* __restrict__ bl,
    const float* __restrict__ Weih, const float* __restrict__ Wehh,
    const float* __restrict__ beih, const float* __restrict__ behh,
    const float* __restrict__ Wenc, const float* __restrict__ benc,
    float* __restrict__ out) {
  __shared__ float  h_lds[NWAVE][16][16];
  __shared__ float4 e_lds[NWAVE][16][12];
  __shared__ float4 x_lds[NWAVE][16][12];

  const int  tid  = threadIdx.x;
  const int  w    = tid >> 5;
  const int  lane = tid & 31;
  const int  m    = lane & 15;
  const bool hi   = lane >= 16;
  const long bb   = (long)blockIdx.x * (NWAVE * BPW) + w * BPW + m;

  // ---------- main LSTM over z ----------
  v16h Aw[4];
  build_A(Wih, Whh, bih, bhh, lane, Aw);

  float hc[8], cc[8];
  #pragma unroll
  for (int r = 0; r < 8; ++r) { hc[r] = 0.0f; cc[r] = 0.0f; }

  const float4* zp = (const float4*)(z + bb * 48);
  for (int t = 0; t < 12; ++t) {
    float4 zt = zp[t];
    v16h bv = build_B(hi, zt.x, zt.y, zt.z, zt.w, hc);
    lstm_step(Aw, bv, cc, hc);
  }

  // stage h_last and the x tile into per-wave LDS
  #pragma unroll
  for (int r = 0; r < 8; ++r) h_lds[w][m][(hi ? 8 : 0) + r] = hc[r];
  {
    const float4* xp = (const float4*)(x + bb * 48);
    const int t0 = hi ? 6 : 0, t1 = hi ? 12 : 6;   // both halves cover batch m
    for (int t = t0; t < t1; ++t) x_lds[w][m][t] = xp[t];
  }
  __syncthreads();

  // ---------- per-batch middle stage: one batch per (low) lane ----------
  if (!hi) {
    float h16[16];
    #pragma unroll
    for (int u = 0; u < 16; ++u) h16[u] = h_lds[w][m][u];

    float m1[8];
    #pragma unroll
    for (int j = 0; j < 8; ++j) {
      float s = mb1[j];
      #pragma unroll
      for (int u = 0; u < 16; ++u) s += h16[u] * mW1[j * 16 + u];
      m1[j] = s > 0.0f ? s : 0.0f;
    }
    float mo[4];
    #pragma unroll
    for (int k = 0; k < 4; ++k) {
      float s = mb2[k];
      #pragma unroll
      for (int j = 0; j < 8; ++j) s += m1[j] * mW2[k * 8 + j];
      mo[k] = s;
    }
    const float g20 = Wg2[0], g21 = Wg2[1], g22 = Wg2[2], g2b = bg2[0];
    const float g10 = Wg[0],  g11 = Wg[1],  g12 = Wg[2],  g1b = bg[0];
    float mog[4];
    gcn4_dev(mo, g20, g21, g22, g2b, mog);

    const float* xr = (const float*)&x_lds[w][m][0];   // x_r[4][12] row-major
    for (int t = 0; t < 12; ++t) {
      float xcol[4], vcol[4];
      const float* wl = Wl + t * 13;
      const float  blt = bl[t];
      #pragma unroll
      for (int nn = 0; nn < 4; ++nn) {
        float s = blt;
        #pragma unroll
        for (int ss = 0; ss < 12; ++ss) s += xr[nn * 12 + ss] * wl[ss];
        s += mog[nn] * wl[12];
        vcol[nn] = s;
        xcol[nn] = xr[nn * 12 + t];
      }
      float e1[4], e2[4];
      gcn4_dev(xcol, g10, g11, g12, g1b, e1);
      gcn4_dev(vcol, g20, g21, g22, g2b, e2);
      e_lds[w][m][t] = make_float4(e1[0] + e2[0], e1[1] + e2[1],
                                   e1[2] + e2[2], e1[3] + e2[3]);
    }
  }
  __syncthreads();

  // ---------- encoder LSTM over en ----------
  v16h Ae[4];
  build_A(Weih, Wehh, beih, behh, lane, Ae);
  #pragma unroll
  for (int r = 0; r < 8; ++r) { hc[r] = 0.0f; cc[r] = 0.0f; }
  for (int t = 0; t < 12; ++t) {
    float4 et = e_lds[w][m][t];
    v16h bv = build_B(hi, et.x, et.y, et.z, et.w, hc);
    lstm_step(Ae, bv, cc, hc);
  }

  #pragma unroll
  for (int r = 0; r < 8; ++r) h_lds[w][m][(hi ? 8 : 0) + r] = hc[r];
  __syncthreads();

  // ---------- last prediction + broadcast-bug output ----------
  if (!hi) {
    float lp[4];
    #pragma unroll
    for (int k = 0; k < 4; ++k) {
      float s = benc[k];
      #pragma unroll
      for (int u = 0; u < 16; ++u) s += h_lds[w][m][u] * Wenc[k * 16 + u];
      lp[k] = s;
    }
    // out[b, t, n] = lp[(t*4 + n) / 12]
    float4* op = (float4*)(out + bb * 48);
    #pragma unroll
    for (int i = 0; i < 12; ++i) {
      float4 o;
      o.x = lp[(4 * i + 0) / 12];
      o.y = lp[(4 * i + 1) / 12];
      o.z = lp[(4 * i + 2) / 12];
      o.w = lp[(4 * i + 3) / 12];
      op[i] = o;
    }
  }
}

extern "C" void kernel_launch(void* const* d_in, const int* in_sizes, int n_in,
                              void* d_out, int out_size, void* d_ws, size_t ws_size,
                              hipStream_t stream) {
  (void)n_in; (void)d_ws; (void)ws_size; (void)out_size;
  const float* x    = (const float*)d_in[0];
  const float* z    = (const float*)d_in[1];
  const float* Wih  = (const float*)d_in[2];
  const float* Whh  = (const float*)d_in[3];
  const float* bih  = (const float*)d_in[4];
  const float* bhh  = (const float*)d_in[5];
  const float* mW1  = (const float*)d_in[6];
  const float* mb1  = (const float*)d_in[7];
  const float* mW2  = (const float*)d_in[8];
  const float* mb2  = (const float*)d_in[9];
  const float* Wg   = (const float*)d_in[10];
  const float* bg   = (const float*)d_in[11];
  const float* Wg2  = (const float*)d_in[12];
  const float* bg2  = (const float*)d_in[13];
  const float* Wl   = (const float*)d_in[14];
  const float* bl   = (const float*)d_in[15];
  const float* Weih = (const float*)d_in[16];
  const float* Wehh = (const float*)d_in[17];
  const float* beih = (const float*)d_in[18];
  const float* behh = (const float*)d_in[19];
  const float* Wenc = (const float*)d_in[20];
  const float* benc = (const float*)d_in[21];

  const int Btot   = in_sizes[0] / 48;            // B = 131072
  const int blocks = Btot / (NWAVE * BPW);        // 1024 blocks of 256 threads

  stae_predict_kernel<<<dim3(blocks), dim3(256), 0, stream>>>(
      x, z, Wih, Whh, bih, bhh, mW1, mb1, mW2, mb2, Wg, bg, Wg2, bg2,
      Wl, bl, Weih, Wehh, beih, behh, Wenc, benc, (float*)d_out);
}
